// GCNFeatureExtractor_35184372089556
// MI455X (gfx1250) — matrix operand order repaired and checked by reading, not compiled, not run on hardware
//
#include <hip/hip_runtime.h>

// Problem constants (from reference): N=8192, E=262144, F_IN=128, HID=256, OUT=128, B=64
#define NNODE  8192
#define NEDGE  262144
#define FIN    128
#define HIDDEN 256
#define FOUT   128
#define NGRAPH 64

typedef __attribute__((ext_vector_type(16))) __bf16        v16bf;
typedef __attribute__((ext_vector_type(8)))  float         v8f;
typedef __attribute__((ext_vector_type(8)))  unsigned int  v8u;

// ---------- bf16 helpers (bit-level, round-to-nearest-even) ----------
__device__ __forceinline__ unsigned short f2bf(float f) {
  unsigned u = __builtin_bit_cast(unsigned, f);
  u += 0x7FFFu + ((u >> 16) & 1u);
  return (unsigned short)(u >> 16);
}
__device__ __forceinline__ float bf2f(unsigned short h) {
  unsigned u = ((unsigned)h) << 16;
  return __builtin_bit_cast(float, u);
}
__device__ __forceinline__ unsigned packbf(float lo, float hi) {
  return (unsigned)f2bf(lo) | ((unsigned)f2bf(hi) << 16);
}

// ---------- fragment loads (pure b128 traffic) ----------
// A fragment (16x32 bf16, lane = row): VGPR0..3 = K kb..kb+7 contiguous,
// VGPR4..7 = K kb+16..kb+23 contiguous  ->  two global_load_b128.
__device__ __forceinline__ v16bf load_frag_a(const unsigned* arow, int kb) {
  const uint4* ap = (const uint4*)(arow + (kb >> 1));   // 16B aligned
  uint4 a0 = ap[0];
  uint4 a1 = ap[2];                                     // +16 elements = +2 uint4
  v8u u;
  u[0] = a0.x; u[1] = a0.y; u[2] = a0.z; u[3] = a0.w;
  u[4] = a1.x; u[5] = a1.y; u[6] = a1.z; u[7] = a1.w;
  return __builtin_bit_cast(v16bf, u);
}
// B fragment (32x16 bf16, lane = col, K-major operand): 16 contiguous K values.
__device__ __forceinline__ v16bf load_frag_b(const unsigned short* p) {
  v8u u = *(const v8u*)p;                               // 32B aligned -> 2x b128
  return __builtin_bit_cast(v16bf, u);
}

// ---------- adjacency build ----------
__global__ void k_zero16(uint4* p, size_t n) {
  size_t i = (size_t)blockIdx.x * blockDim.x + threadIdx.x;
  if (i < n) { uint4 z; z.x = z.y = z.z = z.w = 0u; p[i] = z; }
}

__global__ void k_diag(unsigned short* A) {
  int i = blockIdx.x * blockDim.x + threadIdx.x;
  if (i < NNODE) A[(size_t)i * NNODE + i] = 0x3F80;     // bf16 1.0 (self loop)
}

__global__ void k_edges(unsigned short* A, const long long* ei) {
  int e = blockIdx.x * blockDim.x + threadIdx.x;
  if (e < NEDGE) {
    long long s = ei[e];
    long long d = ei[NEDGE + e];
    // idempotent "set" stores: duplicate edges dedup for free
    A[(size_t)s * NNODE + d] = 0x3F80;
    A[(size_t)d * NNODE + s] = 0x3F80;
  }
}

__device__ __forceinline__ float sumpair(unsigned p) {
  return bf2f((unsigned short)(p & 0xFFFFu)) + bf2f((unsigned short)(p >> 16));
}

// one wave per row: dis[i] = rsqrt(sum_j A[i][j])  (deg>=1 via self loop)
__global__ void k_deg(const unsigned short* __restrict__ A, float* __restrict__ dis) {
  int lane = threadIdx.x & 31;
  int row  = blockIdx.x * 8 + (threadIdx.x >> 5);
  const uint4* rp = (const uint4*)(A + (size_t)row * NNODE);
  float s = 0.f;
  for (int it = lane; it < NNODE / 8; it += 32) {
    uint4 q = rp[it];
    s += sumpair(q.x) + sumpair(q.y) + sumpair(q.z) + sumpair(q.w);
  }
  for (int o = 16; o > 0; o >>= 1) s += __shfl_xor(s, o, 32);
  if (lane == 0) dis[row] = rsqrtf(s);
}

__device__ __forceinline__ unsigned scalepair(unsigned p, float di, const float* dis, int c) {
  float lo = bf2f((unsigned short)(p & 0xFFFFu)) * di * dis[c];
  float hi = bf2f((unsigned short)(p >> 16))     * di * dis[c + 1];
  return packbf(lo, hi);
}

// A[i][j] *= dis[i]*dis[j], 8 elems/thread
__global__ void k_norm(unsigned short* A, const float* __restrict__ dis) {
  size_t i = (size_t)blockIdx.x * blockDim.x + threadIdx.x;
  if (i >= (size_t)NNODE * NNODE / 8) return;
  int row = (int)(i >> 10);                 // 1024 uint4 per row
  int c0  = ((int)i & 1023) << 3;
  float di = dis[row];
  uint4* p = (uint4*)A;
  uint4 q = p[i];
  q.x = scalepair(q.x, di, dis, c0 + 0);
  q.y = scalepair(q.y, di, dis, c0 + 2);
  q.z = scalepair(q.z, di, dis, c0 + 4);
  q.w = scalepair(q.w, di, dis, c0 + 6);
  p[i] = q;
}

// ---------- one-time operand conversions (keeps GEMM loops conversion-free) ----------
__global__ void k_cvt_bf16(const float* __restrict__ src, unsigned short* __restrict__ dst, int n) {
  int i = blockIdx.x * blockDim.x + threadIdx.x;
  if (i < n) dst[i] = f2bf(src[i]);
}
// W (K x NC fp32, row-major) -> Wt (NC x K bf16, K-major)
__global__ void k_wt(const float* __restrict__ W, unsigned short* __restrict__ Wt, int K, int NC) {
  int i = blockIdx.x * blockDim.x + threadIdx.x;
  if (i < K * NC) {
    int k = i / NC, n = i - k * NC;
    Wt[(size_t)n * K + k] = f2bf(W[i]);
  }
}

// ---------- unified WMMA GEMM:  C = op(A(bf16, NNODE x K) @ Bt^T [+ bias]) ----------
// Bt: (NC x K) bf16 K-major.  Output: OUTT -> transposed bf16 (NC x NNODE);
// else row-major (NNODE x NC) as fp32 (OUTF32) or bf16.
// grid = NNODE/16 blocks x 256 threads (8 waves); wave covers TNT 16-col tiles.
template<int K, int NC, int TNT, bool BIAS, bool RELU, bool OUTF32, bool OUTT>
__global__ __launch_bounds__(256) void k_gemm(const unsigned short* __restrict__ A,
                                              const unsigned short* __restrict__ Bt,
                                              const float* __restrict__ bias,
                                              void* __restrict__ C) {
  const int lane = threadIdx.x & 31;
  const int wave = threadIdx.x >> 5;
  const int m0   = blockIdx.x * 16;
  const int n0   = wave * (TNT * 16);
  const int nl   = lane & 15;
  const int mr   = m0 + nl;
  const int hi   = lane >> 4;

  v8f acc[TNT];
  const v8f vz = {0.f, 0.f, 0.f, 0.f, 0.f, 0.f, 0.f, 0.f};
#pragma unroll
  for (int t = 0; t < TNT; ++t) acc[t] = vz;

  const unsigned* arow = (const unsigned*)(A + (size_t)mr * K);

  for (int kk = 0; kk < K; kk += 32) {
    if (K > 2048)   // only the streaming 16KB adjacency rows benefit
      __builtin_prefetch((const void*)(arow + ((kk + 128) >> 1)), 0, 0);

    v16bf af = load_frag_a(arow, kk + hi * 8);
#pragma unroll
    for (int t = 0; t < TNT; ++t) {
      int n = n0 + t * 16 + nl;
      v16bf bf = load_frag_a((const unsigned*)(Bt + (size_t)n * K), kk + hi * 8);
      acc[t] = __builtin_amdgcn_wmma_f32_16x16x32_bf16(false, af, false, bf,
                                                       (short)0, acc[t], false, false);
    }
  }

#pragma unroll
  for (int t = 0; t < TNT; ++t) {
    int n = n0 + t * 16 + nl;
    float bv = BIAS ? bias[n] : 0.f;
#pragma unroll
    for (int v = 0; v < 8; ++v) {
      int m = m0 + v + 8 * hi;
      float val = acc[t][v] + bv;
      if (RELU) val = fmaxf(val, 0.f);
      if (OUTT)        ((unsigned short*)C)[(size_t)n * NNODE + m] = f2bf(val);
      else if (OUTF32) ((float*)C)[(size_t)m * NC + n] = val;
      else             ((unsigned short*)C)[(size_t)m * NC + n] = f2bf(val);
    }
  }
}

// ---------- segmented mean pool over sorted batch ids ----------
__global__ void k_pool(const float* __restrict__ H2, const long long* __restrict__ batch,
                       float* __restrict__ out) {
  int b = blockIdx.x;       // 64 graphs
  int f = threadIdx.x;      // 128 features
  int lo = 0, hiB = NNODE;
  while (lo < hiB) { int mid = (lo + hiB) >> 1; if (batch[mid] < (long long)b) lo = mid + 1; else hiB = mid; }
  int s = lo;
  lo = s; hiB = NNODE;
  while (lo < hiB) { int mid = (lo + hiB) >> 1; if (batch[mid] < (long long)(b + 1)) lo = mid + 1; else hiB = mid; }
  int e = lo;
  float sum = 0.f;
  for (int i = s; i < e; ++i) sum += H2[(size_t)i * FOUT + f];
  out[b * FOUT + f] = sum / fmaxf((float)(e - s), 1.0f);
}

// ---------- launch ----------
extern "C" void kernel_launch(void* const* d_in, const int* in_sizes, int n_in,
                              void* d_out, int out_size, void* d_ws, size_t ws_size,
                              hipStream_t stream) {
  const float*     X     = (const float*)d_in[0];       // (8192,128)
  const float*     W1    = (const float*)d_in[1];       // (128,256)
  const float*     b1    = (const float*)d_in[2];       // (256,)
  const float*     W2    = (const float*)d_in[3];       // (256,128)
  const float*     b2    = (const float*)d_in[4];       // (128,)
  const long long* eidx  = (const long long*)d_in[5];   // (2,262144) int64
  const long long* batch = (const long long*)d_in[6];   // (8192,) int64 sorted
  float*           out   = (float*)d_out;               // (64,128)

  // workspace carve:
  // Adj 128MB | dis 32KB | Xbf 2MB | W1t 64KB | W2t 64KB | XWt 4MB | H 4MB | HWt 2MB | H2 4MB
  char* w = (char*)d_ws;
  unsigned short* Adj = (unsigned short*)w; w += (size_t)NNODE * NNODE * 2;
  float*          dis = (float*)w;          w += (size_t)NNODE * 4;
  unsigned short* Xbf = (unsigned short*)w; w += (size_t)NNODE * FIN * 2;
  unsigned short* W1t = (unsigned short*)w; w += (size_t)HIDDEN * FIN * 2;
  unsigned short* W2t = (unsigned short*)w; w += (size_t)FOUT * HIDDEN * 2;
  unsigned short* XWt = (unsigned short*)w; w += (size_t)HIDDEN * NNODE * 2;
  unsigned short* H   = (unsigned short*)w; w += (size_t)NNODE * HIDDEN * 2;
  unsigned short* HWt = (unsigned short*)w; w += (size_t)FOUT * NNODE * 2;
  float*          H2  = (float*)w;          w += (size_t)NNODE * FOUT * 4;

  // 1) build normalized adjacency (dense; set-stores dedup duplicate edges)
  size_t n16 = (size_t)NNODE * NNODE * 2 / 16;
  k_zero16<<<(unsigned)((n16 + 255) / 256), 256, 0, stream>>>((uint4*)Adj, n16);
  k_diag  <<<NNODE / 256, 256, 0, stream>>>(Adj);
  k_edges <<<NEDGE / 256, 256, 0, stream>>>(Adj, eidx);
  k_deg   <<<NNODE / 8, 256, 0, stream>>>(Adj, dis);
  k_norm  <<<(unsigned)(((size_t)NNODE * NNODE / 8 + 255) / 256), 256, 0, stream>>>(Adj, dis);

  // 2) one-time operand conversions (no conversion left inside GEMM loops)
  k_cvt_bf16<<<(NNODE * FIN + 255) / 256, 256, 0, stream>>>(X, Xbf, NNODE * FIN);
  k_wt<<<(FIN * HIDDEN + 255) / 256, 256, 0, stream>>>(W1, W1t, FIN, HIDDEN);
  k_wt<<<(HIDDEN * FOUT + 255) / 256, 256, 0, stream>>>(W2, W2t, HIDDEN, FOUT);

  // 3) XWt = (X @ W1 + b1)^T
  k_gemm<FIN, HIDDEN, 2, true, false, false, true>
      <<<NNODE / 16, 256, 0, stream>>>(Xbf, W1t, b1, (void*)XWt);

  // 4) H = relu(Adj @ XW)   (K=8192 WMMA GEMM; Adj bf16 fits in 192MB L2 for reuse)
  k_gemm<NNODE, HIDDEN, 2, false, true, false, false>
      <<<NNODE / 16, 256, 0, stream>>>(Adj, XWt, nullptr, (void*)H);

  // 5) HWt = (H @ W2 + b2)^T
  k_gemm<HIDDEN, FOUT, 1, true, false, false, true>
      <<<NNODE / 16, 256, 0, stream>>>(H, W2t, b2, (void*)HWt);

  // 6) H2 = Adj @ HW        (fp32 out for pooling)
  k_gemm<NNODE, FOUT, 1, false, false, true, false>
      <<<NNODE / 16, 256, 0, stream>>>(Adj, HWt, nullptr, (void*)H2);

  // 7) per-graph mean pool
  k_pool<<<NGRAPH, FOUT, 0, stream>>>(H2, batch, out);
}
